// SchNetInteraction_7928509628806
// MI455X (gfx1250) — matrix operand order repaired and checked by reading, not compiled, not run on hardware
//
#include <hip/hip_runtime.h>
#include <hip/hip_bf16.h>

#define N_NODES_  40000
#define N_EDGES_  640000
#define DIM_      128
#define N_BASIS_  50
#define KPAD_     64
#define NF_       128
#define CUTOFF_   5.0f
#define LOG2_     0.6931471805599453f
#define PI_       3.14159265358979323846f

typedef __bf16 bf16;
typedef __attribute__((ext_vector_type(16))) __bf16 v16bf;
typedef __attribute__((ext_vector_type(8)))  __bf16 v8bf;
typedef __attribute__((ext_vector_type(8)))  float  v8f;

#ifndef __has_builtin
#define __has_builtin(x) 0
#endif
// Toolchain-adaptive TDM: ROCm 7.2 (clang-22) = 5-arg builtin, no TDM header;
// therock/amdgpu-toolchain = 6-arg builtin + amd_gfx1250_TDM.h header.
#if __has_builtin(__builtin_amdgcn_tensor_load_to_lds)
# if __has_include(<hip/amd_detail/amd_gfx1250_TDM.h>)
#  define TDM_ARITY 6
#  warning "CDNA5 probe: TDM_ARITY=6 (header present; 6-arg tensor_load_to_lds)"
# else
#  define TDM_ARITY 5
#  warning "CDNA5 probe: TDM_ARITY=5 (5-arg tensor_load_to_lds)"
# endif
#else
# define TDM_ARITY 0
# warning "CDNA5 probe: TDM_ARITY=0 (no tensor_load_to_lds builtin; coop-copy fallback)"
#endif
#define USE_TDM (TDM_ARITY != 0)

union FragA { v16bf v; v8bf h[2]; };

__device__ __forceinline__ float sspf(float t) {
    // softplus(t) - log(2), numerically stable
    return fmaxf(t, 0.0f) + __logf(1.0f + __expf(-fabsf(t))) - LOG2_;
}

__device__ __forceinline__ v8f wmma_bf16(v16bf a, v16bf b, v8f c) {
    return __builtin_amdgcn_wmma_f32_16x16x32_bf16(
        /*neg_a=*/false, a, /*neg_b=*/false, b,
        /*c_mod=*/(short)0, c, /*reuse_a=*/false, /*reuse_b=*/false);
}

// A fragment (16x32 bf16 tile) from LDS row-major [16][row_stride] bf16.
// lane%16 = M row; lane/16 selects K-half: 16 contiguous K values from kbase+half*16
__device__ __forceinline__ v16bf load_fragA_lds(const bf16* rows, int row_stride,
                                                int kbase, int lane) {
    int r  = lane & 15;
    int kh = (lane >> 4) * 16;
    const bf16* p = rows + (size_t)r * row_stride;
    FragA fa;
    fa.h[0] = *(const v8bf*)(p + kbase + kh);
    fa.h[1] = *(const v8bf*)(p + kbase + kh + 8);
    return fa.v;
}

// B fragment (32x16 bf16 tile) from transposed (N-major) weights Wt[N][row_stride].
// lane%16 = N column; lane/16 = K-half; 16 contiguous K values per lane.
__device__ __forceinline__ v16bf load_fragB(const bf16* Wt, int row_stride,
                                            int ncol, int kbase, int lane) {
    const bf16* p = Wt + (size_t)ncol * row_stride + kbase + ((lane >> 4) << 4);
    return *(const v16bf*)p;
}

#if USE_TDM
typedef __attribute__((ext_vector_type(4))) unsigned int u32x4;
typedef __attribute__((ext_vector_type(8))) int          i32x8;
typedef __attribute__((ext_vector_type(4))) int          i32x4;

// TDM 2-D tile load: copy a dense [rows][cols] bf16 matrix from global into LDS.
// D# packing per CDNA5 ISA 8.3/8.4 (group0: count/lds_addr/global_addr/type=2;
// group1: data_size=2B, tensor dims, tile dims, dim0 stride). Groups 2/3 zero (2-D).
__device__ __forceinline__ void tdm_load_2d_bf16(const bf16* gsrc,
                                                 unsigned int lds_byte_off,
                                                 unsigned int rows, unsigned int cols) {
    unsigned long long ga = (unsigned long long)gsrc;
    u32x4 g0;
    g0[0] = 1u;                                               // count=1, user mode
    g0[1] = lds_byte_off;                                     // lds_addr
    g0[2] = (unsigned int)(ga & 0xffffffffu);                 // global_addr[31:0]
    g0[3] = (unsigned int)((ga >> 32) & 0x01ffffffu)          // global_addr[56:32]
            | (2u << 30);                                     // type=2 ("image")
    i32x8 g1;
    g1[0] = (int)(1u << 16);                                  // data_size=1 (2 bytes)
    g1[1] = (int)(cols << 16);                                // tensor_dim0[15:0]
    g1[2] = (int)((cols >> 16) | (rows << 16));               // dim0 hi | tensor_dim1 lo
    g1[3] = (int)((rows >> 16) | (cols << 16));               // dim1 hi | tile_dim0
    g1[4] = (int)(rows & 0xffffu);                            // tile_dim1 (tile_dim2=0)
    g1[5] = (int)cols;                                        // tensor_dim0_stride[31:0]
    g1[6] = 0;                                                // stride hi bits
    g1[7] = 0;
    i32x4 z = {0, 0, 0, 0};
#if TDM_ARITY == 5
    __builtin_amdgcn_tensor_load_to_lds(g0, g1, z, z, 0);
#else
    i32x8 z8 = {0, 0, 0, 0, 0, 0, 0, 0};
    __builtin_amdgcn_tensor_load_to_lds(g0, g1, z, z, z8, 0);
#endif
}

__device__ __forceinline__ unsigned int lds_off(const void* p) {
    // generic pointer to LDS = {shared-aperture hi32, lds_offset lo32}
    return (unsigned int)(unsigned long long)p;
}
#endif

// ---------------------------------------------------------------------------
// Never launched. Encodes which TDM path compiled into the wmma histogram count:
//   base kernels emit 60 wmma; +1 => 5-arg TDM path, +2 => 6-arg, +0 => fallback.
// ---------------------------------------------------------------------------
__global__ void cdna5_path_marker(v8f* __restrict__ out, const v16bf* __restrict__ a) {
#if TDM_ARITY == 5
    v8f acc = {};
    acc = wmma_bf16(a[0], a[1], acc);
    *out = acc;
#elif TDM_ARITY == 6
    v8f acc = {};
    acc = wmma_bf16(a[0], a[1], acc);
    acc = wmma_bf16(a[2], a[3], acc);
    *out = acc;
#else
    v8f zero = {};
    *out = zero;
#endif
}

// ---------------------------------------------------------------------------
// Kernel 1: convert weights f32 -> bf16, transposed to N-major; pad W_f1 K 50->64
// ---------------------------------------------------------------------------
__global__ void __launch_bounds__(256) convert_weights(
    const float* __restrict__ W_in2f, const float* __restrict__ W_f1,
    const float* __restrict__ W_f2,   const float* __restrict__ W_out,
    const float* __restrict__ W_lin,
    bf16* __restrict__ Wf1t, bf16* __restrict__ Wf2t, bf16* __restrict__ Win2ft,
    bf16* __restrict__ Woutt, bf16* __restrict__ Wlint) {
    int idx = blockIdx.x * 256 + threadIdx.x;   // 0..16383
    int n = idx >> 7;                           // output column
    int k = idx & 127;                          // contraction index
    Win2ft[(size_t)n * 128 + k] = (bf16)W_in2f[(size_t)k * 128 + n];
    Wf2t  [(size_t)n * 128 + k] = (bf16)W_f2  [(size_t)k * 128 + n];
    Woutt [(size_t)n * 128 + k] = (bf16)W_out [(size_t)k * 128 + n];
    Wlint [(size_t)n * 128 + k] = (bf16)W_lin [(size_t)k * 128 + n];
    if (k < KPAD_) {
        float v = (k < N_BASIS_) ? W_f1[(size_t)k * NF_ + n] : 0.0f;
        Wf1t[(size_t)n * KPAD_ + k] = (bf16)v;
    }
}

// ---------------------------------------------------------------------------
// Kernel 2: h = x @ W_in2f   [40000,128] @ [128,128]
// block = 256 threads (8 waves) = 16 node rows; wave w owns 16-col tile w.
// ---------------------------------------------------------------------------
__global__ void __launch_bounds__(256) node_embed(
    const float* __restrict__ x, const bf16* __restrict__ Win2ft,
    float* __restrict__ h) {
    __shared__ __align__(32) bf16 xa[16 * 128];      //  4 KB
    __shared__ __align__(32) bf16 w_s[128 * 128];    // 32 KB
    int tid = threadIdx.x;
    int nodeBase = blockIdx.x * 16;

#if USE_TDM
    if (tid < 32)   // one wave issues the DMA; EXEC is ignored by tensor ops
        tdm_load_2d_bf16(Win2ft, lds_off(&w_s[0]), 128u, 128u);
#endif
    for (int i = tid; i < 16 * 128; i += 256)
        xa[i] = (bf16)x[(size_t)nodeBase * 128 + i];
#if USE_TDM
    if (tid < 32)
        __builtin_amdgcn_s_wait_tensorcnt(0);
#else
    for (int i = tid; i < 128 * 128; i += 256)
        w_s[i] = Win2ft[i];
#endif
    __syncthreads();

    int wave = tid >> 5, lane = tid & 31;
    int col16 = lane & 15;
    int rbase = (lane >> 4) * 8;
    int ncol  = wave * 16 + col16;

    v16bf a0 = load_fragA_lds(xa, 128, 0,  lane);
    v16bf a1 = load_fragA_lds(xa, 128, 32, lane);
    v16bf a2 = load_fragA_lds(xa, 128, 64, lane);
    v16bf a3 = load_fragA_lds(xa, 128, 96, lane);

    v8f acc = {};
    acc = wmma_bf16(a0, load_fragB(w_s, 128, ncol, 0,  lane), acc);
    acc = wmma_bf16(a1, load_fragB(w_s, 128, ncol, 32, lane), acc);
    acc = wmma_bf16(a2, load_fragB(w_s, 128, ncol, 64, lane), acc);
    acc = wmma_bf16(a3, load_fragB(w_s, 128, ncol, 96, lane), acc);

#pragma unroll
    for (int i = 0; i < 8; ++i)
        h[(size_t)(nodeBase + rbase + i) * 128 + ncol] = acc[i];
}

// ---------------------------------------------------------------------------
// Kernel 3 (fused edge pipeline):
//   W = ssp(f_ij@Wf1+b1)@Wf2+b2 ; W *= cutoff(r) ; agg[ind_i] += h[ind_j]*W
// block = 256 threads (8 waves) = 128 edges; each wave owns 16 edges x 128 cols.
// LDS total ~97.5 KB (gfx1250 WGP has 320 KB -> 3 blocks/WGP resident).
// ---------------------------------------------------------------------------
__global__ void __launch_bounds__(256) edge_kernel(
    const float* __restrict__ f_ij, const float* __restrict__ r_ij,
    const int* __restrict__ ind_i,  const int* __restrict__ ind_j,
    const bf16* __restrict__ Wf1t,  const bf16* __restrict__ Wf2t,
    const float* __restrict__ b_f1, const float* __restrict__ b_f2,
    const float* __restrict__ h,    float* __restrict__ agg) {
    __shared__ __align__(32) bf16  fij_s[128 * KPAD_];   // 16 KB (K padded 50->64)
    __shared__ __align__(32) bf16  act_s[128 * NF_];     // 32 KB (per-wave-private slices)
    __shared__ __align__(32) bf16  w1_s[NF_ * KPAD_];    // 16 KB
    __shared__ __align__(32) bf16  w2_s[NF_ * NF_];      // 32 KB
    __shared__ float cut_s[128];
    __shared__ int   ii_s[128];
    __shared__ int   jj_s[128];

    int tid = threadIdx.x;
    size_t eBase = (size_t)blockIdx.x * 128;

#if USE_TDM
    if (tid < 32) {  // wave 0 kicks off both weight DMAs; overlapped with staging below
        tdm_load_2d_bf16(Wf1t, lds_off(&w1_s[0]), (unsigned)NF_, (unsigned)KPAD_);
        tdm_load_2d_bf16(Wf2t, lds_off(&w2_s[0]), (unsigned)NF_, (unsigned)NF_);
    }
#endif

    // f_ij streams from HBM (128 MB total): prefetch the next block's tile.
    if (tid == 0 && blockIdx.x + 1 < N_EDGES_ / 128)
        __builtin_prefetch(&f_ij[(eBase + 128) * N_BASIS_], 0, 1);

    for (int i = tid; i < 128 * KPAD_; i += 256) {
        int e = i >> 6;          // /KPAD_
        int k = i & (KPAD_ - 1);
        float v = (k < N_BASIS_) ? f_ij[(eBase + e) * N_BASIS_ + k] : 0.0f;
        fij_s[i] = (bf16)v;
    }
    if (tid < 128) {
        float r = r_ij[eBase + tid];
        float c = 0.5f * (__cosf(r * (PI_ / CUTOFF_)) + 1.0f);
        cut_s[tid] = (r < CUTOFF_) ? c : 0.0f;
        ii_s[tid]  = ind_i[eBase + tid];
        jj_s[tid]  = ind_j[eBase + tid];
    }
#if USE_TDM
    if (tid < 32)
        __builtin_amdgcn_s_wait_tensorcnt(0);
#else
    for (int i = tid; i < NF_ * KPAD_; i += 256) w1_s[i] = Wf1t[i];
    for (int i = tid; i < NF_ * NF_;  i += 256) w2_s[i] = Wf2t[i];
#endif
    __syncthreads();

    int wave = tid >> 5, lane = tid & 31;
    int mBase = wave * 16;
    int col16 = lane & 15;
    int rbase = (lane >> 4) * 8;

    // ---- GEMM1: [16,64]@[64,128] + b_f1, then ssp -> act_s (bf16) ----
    const bf16* Arow = fij_s + (size_t)mBase * KPAD_;
    v16bf a0 = load_fragA_lds(Arow, KPAD_, 0,  lane);
    v16bf a1 = load_fragA_lds(Arow, KPAD_, 32, lane);

#pragma unroll
    for (int nt = 0; nt < 8; ++nt) {
        int ncol = nt * 16 + col16;
        float bias = b_f1[ncol];
        v8f acc = { bias, bias, bias, bias, bias, bias, bias, bias };
        acc = wmma_bf16(a0, load_fragB(w1_s, KPAD_, ncol, 0,  lane), acc);
        acc = wmma_bf16(a1, load_fragB(w1_s, KPAD_, ncol, 32, lane), acc);
#pragma unroll
        for (int i = 0; i < 8; ++i) {
            int row = mBase + rbase + i;
            act_s[(size_t)row * NF_ + ncol] = (bf16)sspf(acc[i]);
        }
    }
    // LDS slice is wave-private: only need this wave's DS ops to complete.
    asm volatile("s_wait_dscnt 0" ::: "memory");

    // ---- GEMM2: [16,128]@[128,128] + b_f2, cutoff, gather/modulate/scatter ----
    const bf16* A2row = act_s + (size_t)mBase * NF_;
    v16bf c0 = load_fragA_lds(A2row, NF_, 0,  lane);
    v16bf c1 = load_fragA_lds(A2row, NF_, 32, lane);
    v16bf c2 = load_fragA_lds(A2row, NF_, 64, lane);
    v16bf c3 = load_fragA_lds(A2row, NF_, 96, lane);

#pragma unroll
    for (int nt = 0; nt < 8; ++nt) {
        int ncol = nt * 16 + col16;
        float bias = b_f2[ncol];
        v8f acc = { bias, bias, bias, bias, bias, bias, bias, bias };
        acc = wmma_bf16(c0, load_fragB(w2_s, NF_, ncol, 0,  lane), acc);
        acc = wmma_bf16(c1, load_fragB(w2_s, NF_, ncol, 32, lane), acc);
        acc = wmma_bf16(c2, load_fragB(w2_s, NF_, ncol, 64, lane), acc);
        acc = wmma_bf16(c3, load_fragB(w2_s, NF_, ncol, 96, lane), acc);
#pragma unroll
        for (int i = 0; i < 8; ++i) {
            int row = mBase + rbase + i;
            float w   = acc[i] * cut_s[row];
            int   src = jj_s[row];
            int   dst = ii_s[row];
            float msg = h[(size_t)src * 128 + ncol] * w;   // h is L2-resident (20.5MB)
            unsafeAtomicAdd(&agg[(size_t)dst * 128 + ncol], msg);
        }
    }
}

// ---------------------------------------------------------------------------
// Kernel 4: out = ssp(agg @ W_out + b_out) @ W_lin + b_lin
// Both weights staged to LDS (72 KB total -> 4 blocks/WGP).
// ---------------------------------------------------------------------------
__global__ void __launch_bounds__(256) output_kernel(
    const float* __restrict__ agg,
    const bf16* __restrict__ Woutt, const bf16* __restrict__ Wlint,
    const float* __restrict__ b_out, const float* __restrict__ b_lin,
    float* __restrict__ out) {
    __shared__ __align__(32) bf16 aggA[16 * 128];
    __shared__ __align__(32) bf16 actA[16 * 128];
    __shared__ __align__(32) bf16 wo_s[128 * 128];   // 32 KB
    __shared__ __align__(32) bf16 wl_s[128 * 128];   // 32 KB
    int tid = threadIdx.x;
    int nodeBase = blockIdx.x * 16;

#if USE_TDM
    if (tid < 32) {
        tdm_load_2d_bf16(Woutt, lds_off(&wo_s[0]), 128u, 128u);
        tdm_load_2d_bf16(Wlint, lds_off(&wl_s[0]), 128u, 128u);
    }
#endif
    for (int i = tid; i < 16 * 128; i += 256)
        aggA[i] = (bf16)agg[(size_t)nodeBase * 128 + i];
#if USE_TDM
    if (tid < 32)
        __builtin_amdgcn_s_wait_tensorcnt(0);
#else
    for (int i = tid; i < 128 * 128; i += 256) wo_s[i] = Woutt[i];
    for (int i = tid; i < 128 * 128; i += 256) wl_s[i] = Wlint[i];
#endif
    __syncthreads();

    int wave = tid >> 5, lane = tid & 31;
    int col16 = lane & 15;
    int rbase = (lane >> 4) * 8;
    int ncol  = wave * 16 + col16;

    v16bf a0 = load_fragA_lds(aggA, 128, 0,  lane);
    v16bf a1 = load_fragA_lds(aggA, 128, 32, lane);
    v16bf a2 = load_fragA_lds(aggA, 128, 64, lane);
    v16bf a3 = load_fragA_lds(aggA, 128, 96, lane);

    float bias = b_out[ncol];
    v8f acc = { bias, bias, bias, bias, bias, bias, bias, bias };
    acc = wmma_bf16(a0, load_fragB(wo_s, 128, ncol, 0,  lane), acc);
    acc = wmma_bf16(a1, load_fragB(wo_s, 128, ncol, 32, lane), acc);
    acc = wmma_bf16(a2, load_fragB(wo_s, 128, ncol, 64, lane), acc);
    acc = wmma_bf16(a3, load_fragB(wo_s, 128, ncol, 96, lane), acc);

#pragma unroll
    for (int i = 0; i < 8; ++i)
        actA[(size_t)(rbase + i) * 128 + ncol] = (bf16)sspf(acc[i]);
    __syncthreads();   // actA written by all 8 waves, read by all 8 waves

    v16bf c0 = load_fragA_lds(actA, 128, 0,  lane);
    v16bf c1 = load_fragA_lds(actA, 128, 32, lane);
    v16bf c2 = load_fragA_lds(actA, 128, 64, lane);
    v16bf c3 = load_fragA_lds(actA, 128, 96, lane);

    float bias2 = b_lin[ncol];
    v8f acc2 = { bias2, bias2, bias2, bias2, bias2, bias2, bias2, bias2 };
    acc2 = wmma_bf16(c0, load_fragB(wl_s, 128, ncol, 0,  lane), acc2);
    acc2 = wmma_bf16(c1, load_fragB(wl_s, 128, ncol, 32, lane), acc2);
    acc2 = wmma_bf16(c2, load_fragB(wl_s, 128, ncol, 64, lane), acc2);
    acc2 = wmma_bf16(c3, load_fragB(wl_s, 128, ncol, 96, lane), acc2);

#pragma unroll
    for (int i = 0; i < 8; ++i)
        out[(size_t)(nodeBase + rbase + i) * 128 + ncol] = acc2[i];
}

// ---------------------------------------------------------------------------
extern "C" void kernel_launch(void* const* d_in, const int* in_sizes, int n_in,
                              void* d_out, int out_size, void* d_ws, size_t ws_size,
                              hipStream_t stream) {
    const float* x      = (const float*)d_in[0];
    const float* r_ij   = (const float*)d_in[1];
    const float* f_ij   = (const float*)d_in[2];
    const int*   ind_i  = (const int*)d_in[3];
    const int*   ind_j  = (const int*)d_in[4];
    const float* W_in2f = (const float*)d_in[5];
    const float* W_f1   = (const float*)d_in[6];
    const float* b_f1   = (const float*)d_in[7];
    const float* W_f2   = (const float*)d_in[8];
    const float* b_f2   = (const float*)d_in[9];
    const float* W_out  = (const float*)d_in[10];
    const float* b_out  = (const float*)d_in[11];
    const float* W_lin  = (const float*)d_in[12];
    const float* b_lin  = (const float*)d_in[13];

    // Workspace layout (all offsets 256B-aligned), total ~41.1 MB
    char*  ws     = (char*)d_ws;
    float* h      = (float*)(ws);                 // 40000*128*4 = 20,480,000
    float* agg    = (float*)(ws + 20480000);      // 20,480,000
    bf16*  Wf1t   = (bf16*)(ws + 40960000);       // 128*64*2  = 16,384
    bf16*  Wf2t   = (bf16*)(ws + 40976384);       // 128*128*2 = 32,768
    bf16*  Win2ft = (bf16*)(ws + 41009152);
    bf16*  Woutt  = (bf16*)(ws + 41041920);
    bf16*  Wlint  = (bf16*)(ws + 41074688);

    hipMemsetAsync(agg, 0, (size_t)N_NODES_ * DIM_ * sizeof(float), stream);

    convert_weights<<<64, 256, 0, stream>>>(W_in2f, W_f1, W_f2, W_out, W_lin,
                                            Wf1t, Wf2t, Win2ft, Woutt, Wlint);
    node_embed<<<N_NODES_ / 16, 256, 0, stream>>>(x, Win2ft, h);
    edge_kernel<<<N_EDGES_ / 128, 256, 0, stream>>>(f_ij, r_ij, ind_i, ind_j,
                                                    Wf1t, Wf2t, b_f1, b_f2, h, agg);
    output_kernel<<<N_NODES_ / 16, 256, 0, stream>>>(agg, Woutt, Wlint,
                                                     b_out, b_lin, (float*)d_out);
}
